// LSTM_36386962931817
// MI455X (gfx1250) — matrix operand order repaired and checked by reading, not compiled, not run on hardware
//
#include <hip/hip_runtime.h>
#include <hip/hip_bf16.h>
#include <math.h>

// ---------------------------------------------------------------------------
// LSTM scan, T=512, B=256, F=U=128, 4U=512.  (MI455X / gfx1250, wave32)
//
//  * 16 workgroups; each owns a 16-row batch slice (== WMMA M tile) and
//    marches the full 512-step recurrence privately (no global sync).
//  * Wx/Wh converted to bf16 once into LDS, then each wave hoists its 16
//    B-tiles (128 VGPRs) into registers for the whole scan.
//  * Per step: z[16,512] = x@Wx + h@Wh via v_wmma_f32_16x16x32_bf16
//    (16 waves x 2 N-tiles x 8 K-steps); only A operands re-read from LDS.
//  * x_t staged into LDS by the Tensor Data Mover (tensor_load_to_lds,
//    double-buffered 2 steps ahead, s_wait_tensorcnt) when available.
//  * Gates use branch-free v_exp_f32/v_rcp_f32 sigmoid/tanh.
// ---------------------------------------------------------------------------

typedef __attribute__((ext_vector_type(16))) __bf16 v16bf;
typedef __attribute__((ext_vector_type(8)))  float  v8f;
typedef unsigned int __attribute__((ext_vector_type(4))) v4u;
typedef int  __attribute__((ext_vector_type(4))) v4i;
typedef int  __attribute__((ext_vector_type(8))) v8i;

#define T_STEPS 512
#define BATCH   256
#define FEAT    128
#define UNITS   128
#define GATES   512

#define BSLICE  16
#define NTHREADS 512           // 16 wave32 waves

#define WPITCH  516            // words per K-pair row of weight image
#define APITCH  66             // words per row of bf16-pair staging
#define ZPITCH  516            // f32 words per row of z buffer
#define XSPITCH 132            // f32 words per row of TDM x stage (128 + 4 pad)

#define W_WORDS   (64 * WPITCH)
#define A_WORDS   (BSLICE * APITCH)
#define Z_WORDS   (BSLICE * ZPITCH)
#define XS_WORDS  (BSLICE * XSPITCH)

#if __has_builtin(__builtin_amdgcn_tensor_load_to_lds)
#define USE_TDM 1
#else
#define USE_TDM 0
#endif

#if USE_TDM
#define LDS_WORDS (2 * W_WORDS + 2 * A_WORDS + Z_WORDS + 2 * XS_WORDS)
#else
#define LDS_WORDS (2 * W_WORDS + 2 * A_WORDS + Z_WORDS)
#endif
#define LDS_BYTES (LDS_WORDS * 4)      // <= 322,560 B  (< 320 KB WGP LDS)

static __device__ __forceinline__ unsigned short f2bf(float f) {
  unsigned int u = __builtin_bit_cast(unsigned int, f);
  u += 0x7FFFu + ((u >> 16) & 1u);
  return (unsigned short)(u >> 16);
}

// Branch-free gate activations on the hardware transcendental pipe:
// v_exp_f32 (exp2) + v_rcp_f32. Saturates correctly at +-inf.
#if __has_builtin(__builtin_amdgcn_exp2f) && __has_builtin(__builtin_amdgcn_rcpf)
static __device__ __forceinline__ float fast_sigmoid(float x) {
  return __builtin_amdgcn_rcpf(1.0f + __builtin_amdgcn_exp2f(x * -1.442695041f));
}
static __device__ __forceinline__ float fast_tanh(float x) {
  const float r = __builtin_amdgcn_rcpf(1.0f + __builtin_amdgcn_exp2f(x * -2.885390082f));
  return 2.0f * r - 1.0f;
}
#else
static __device__ __forceinline__ float fast_sigmoid(float x) { return 1.0f / (1.0f + expf(-x)); }
static __device__ __forceinline__ float fast_tanh(float x)    { return tanhf(x); }
#endif

// A operand: 16x32 bf16 tile (packed K-pairs, pitch APITCH).
// lanes 0-15 rows M, K=(0..15)+16*khalf contiguous in VGPR0-7 (matches B).
static __device__ __forceinline__ v16bf load_a(const unsigned int* buf,
                                               int m, int khalf, int kb) {
  union { v16bf v; unsigned int u[8]; } A;
  const int kbase = kb * 32 + khalf * 16;
#pragma unroll
  for (int r = 0; r < 8; ++r) {
    const int k = kbase + (r << 1);
    A.u[r] = buf[m * APITCH + (k >> 1)];
  }
  return A.v;
}

// B operand: 32x16 bf16 tile from weight LDS image (loaded ONCE into regs).
static __device__ __forceinline__ v16bf load_b(const unsigned int* W,
                                               int n, int khalf, int kb, int n0) {
  union { v16bf v; unsigned int u[8]; } B;
  const int kpbase = kb * 16 + khalf * 8;
#pragma unroll
  for (int r = 0; r < 8; ++r)
    B.u[r] = W[(kpbase + r) * WPITCH + n0 + n];
  return B.v;
}

#if USE_TDM
// Issue one TDM 2D-tile load: 16 rows x 128 f32, dense row stride 128,
// LDS padded 4 dwords every 128 (pad_interval=6, pad_amount=3 -> pitch 132).
// D# bit layout per CDNA5 ISA ch.8.3/8.4; group0.type=2, count=1, dsize=4B.
static __device__ __forceinline__ void tdm_load_x(const float* gsrc,
                                                  unsigned lds_byte) {
  const unsigned long long ga = (unsigned long long)(uintptr_t)gsrc;
  v4u g0;
  g0[0] = 1u;                                          // count=1, user mode
  g0[1] = lds_byte;                                    // lds_addr
  g0[2] = (unsigned)ga;                                // global_addr[31:0]
  g0[3] = (unsigned)((ga >> 32) & 0x01FFFFFFu) | (2u << 30);  // [56:32] | type=2
  v8i g1;
  g1[0] = (int)((2u << 16) | (1u << 20) | (6u << 22) | (3u << 25));
  //            data_size=4B  pad_en      pad_int=128dw  pad_amt=4dw
  g1[1] = (int)(128u << 16);                           // tensor_dim0 = 128
  g1[2] = (int)(16u << 16);                            // tensor_dim1 = 16
  g1[3] = (int)(128u << 16);                           // tile_dim0 = 128
  g1[4] = 16;                                          // tile_dim1 = 16
  g1[5] = 128;                                         // tensor_dim0_stride
  g1[6] = 0;
  g1[7] = 0;
  v4i z4 = {0, 0, 0, 0};
#if __clang_major__ >= 23
  v8i z8 = {0, 0, 0, 0, 0, 0, 0, 0};
  __builtin_amdgcn_tensor_load_to_lds(g0, g1, z4, z4, z8, 0);
#else
  __builtin_amdgcn_tensor_load_to_lds(g0, g1, z4, z4, 0);
#endif
}

// Wait until TENSORcnt <= 1 (the older of the two in-flight tiles landed).
// SIMM16 must be a compile-time constant.
static __device__ __forceinline__ void wait_tensorcnt_le1() {
#if __has_builtin(__builtin_amdgcn_s_wait_tensorcnt)
  __builtin_amdgcn_s_wait_tensorcnt(1);
#else
  asm volatile("s_wait_tensorcnt 0x1" ::: "memory");
#endif
}
#endif  // USE_TDM

__global__ __launch_bounds__(NTHREADS, 1)
void lstm_scan_wmma(const float* __restrict__ inputs,   // [T,B,F]
                    const float* __restrict__ states,   // [2,B,U]
                    const float* __restrict__ dones,    // [T,B]
                    const float* __restrict__ kernelx,  // [F,4U]
                    const float* __restrict__ kernelh,  // [U,4U]
                    float* __restrict__ out)            // [T,B,U] ++ [2,B,U]
{
  extern __shared__ unsigned int lds[];
  unsigned int*  wx   = lds;
  unsigned int*  wh   = wx + W_WORDS;
  unsigned int*  xbuf = wh + W_WORDS;                // x_t slice, bf16 pairs
  unsigned int*  hbuf = xbuf + A_WORDS;              // h slice, bf16 pairs
  float*         zbuf = (float*)(hbuf + A_WORDS);    // gate pre-activations
#if USE_TDM
  float*         xstage[2] = { zbuf + Z_WORDS, zbuf + Z_WORDS + XS_WORDS };
#endif

  unsigned short* xbuf16 = (unsigned short*)xbuf;
  unsigned short* hbuf16 = (unsigned short*)hbuf;

  const int tid   = threadIdx.x;
  const int lane  = tid & 31;
  const int wave  = tid >> 5;
  const int b0    = blockIdx.x * BSLICE;

#if USE_TDM
  // Prime the TDM pipeline: tiles for t=0 and t=1 (wave 0 issues; EXEC
  // is ignored by tensor ops, the branch is wave-uniform).
  if (wave == 0) {
    tdm_load_x(inputs + (size_t)0 * BATCH * FEAT + (size_t)b0 * FEAT,
               (unsigned)(uintptr_t)(void*)xstage[0]);
    tdm_load_x(inputs + (size_t)1 * BATCH * FEAT + (size_t)b0 * FEAT,
               (unsigned)(uintptr_t)(void*)xstage[1]);
  }
#endif

  // ---- one-time: weight matrices f32 -> bf16-pair LDS image --------------
  for (int w = tid; w < 64 * GATES; w += NTHREADS) {
    const int kp = w >> 9;
    const int n  = w & 511;
    wx[kp * WPITCH + n] =
        (unsigned int)f2bf(kernelx[(2 * kp) * GATES + n]) |
        ((unsigned int)f2bf(kernelx[(2 * kp + 1) * GATES + n]) << 16);
    wh[kp * WPITCH + n] =
        (unsigned int)f2bf(kernelh[(2 * kp) * GATES + n]) |
        ((unsigned int)f2bf(kernelh[(2 * kp + 1) * GATES + n]) << 16);
  }

  // ---- one-time: initial c,h (4 (b,u) pairs per thread) ------------------
  float c_reg[4], h_reg[4];
#pragma unroll
  for (int k = 0; k < 4; ++k) {
    const int idx = tid + k * NTHREADS;
    const int b = idx >> 7, u = idx & 127;
    c_reg[k] = states[(b0 + b) * UNITS + u];
    const float h0 = states[BATCH * UNITS + (b0 + b) * UNITS + u];
    h_reg[k] = h0;
    hbuf16[b * (2 * APITCH) + u] = f2bf(h0);
  }
#if USE_TDM
  if (wave == 0) wait_tensorcnt_le1();   // tile for t=0 landed
#endif
  __syncthreads();

  const int ntile0 = wave * 2;
  const int m      = lane & 15;
  const int khalf  = lane >> 4;

  // ---- hoist all weight B-tiles into VGPRs for the entire scan -----------
  v16bf bwx[2][4], bwh[2][4];
#pragma unroll
  for (int s = 0; s < 2; ++s) {
    const int n0 = (ntile0 + s) * 16;
#pragma unroll
    for (int kb = 0; kb < 4; ++kb) {
      bwx[s][kb] = load_b(wx, m, khalf, kb, n0);
      bwh[s][kb] = load_b(wh, m, khalf, kb, n0);
    }
  }

  for (int t = 0; t < T_STEPS; ++t) {
    // -- (a) stage x_t as bf16; mask c (regs) and h (LDS) on episode reset --
#pragma unroll
    for (int k = 0; k < 4; ++k) {
      const int idx = tid + k * NTHREADS;
      const int b = idx >> 7, f = idx & 127;
#if USE_TDM
      const float xv = xstage[t & 1][b * XSPITCH + f];
#else
      const float xv = inputs[((size_t)t * BATCH + b0 + b) * FEAT + f];
#endif
      xbuf16[b * (2 * APITCH) + f] = f2bf(xv);
      if (dones[t * BATCH + b0 + b] != 0.0f) {
        c_reg[k] = 0.0f;
        hbuf16[b * (2 * APITCH) + (idx & 127)] = 0;
      }
    }
#if !USE_TDM
    if (t + 1 < T_STEPS)
      __builtin_prefetch(&inputs[((size_t)(t + 1) * BATCH + b0 + (tid >> 7)) * FEAT
                                 + (tid & 127)], 0, 3);
#endif
    __syncthreads();

#if USE_TDM
    // xstage[t&1] fully consumed (all ds reads retired before the barrier);
    // refill it with the tile for t+2 while the GEMM + gates run.
    if (wave == 0 && t + 2 < T_STEPS)
      tdm_load_x(inputs + ((size_t)(t + 2) * BATCH + b0) * FEAT,
                 (unsigned)(uintptr_t)(void*)xstage[t & 1]);
#endif

    // -- (c) z[16,512] = x@Wx + h@Wh; A from LDS once, B from registers ----
    v16bf ax[4], ah[4];
#pragma unroll
    for (int kb = 0; kb < 4; ++kb) {
      ax[kb] = load_a(xbuf, m, khalf, kb);
      ah[kb] = load_a(hbuf, m, khalf, kb);
    }
#pragma unroll
    for (int s = 0; s < 2; ++s) {
      const int n0 = (ntile0 + s) * 16;
      v8f acc = {};
#pragma unroll
      for (int kb = 0; kb < 4; ++kb)
        acc = __builtin_amdgcn_wmma_f32_16x16x32_bf16(
            false, ax[kb], false, bwx[s][kb], (short)0, acc, false, false);
#pragma unroll
      for (int kb = 0; kb < 4; ++kb)
        acc = __builtin_amdgcn_wmma_f32_16x16x32_bf16(
            false, ah[kb], false, bwh[s][kb], (short)0, acc, false, false);
#pragma unroll
      for (int r = 0; r < 8; ++r)
        zbuf[(r + 8 * khalf) * ZPITCH + n0 + m] = acc[r];
    }
    __syncthreads();

    // -- (e) gates + state update (branch-free transcendental pipe) --------
#pragma unroll
    for (int k = 0; k < 4; ++k) {
      const int idx = tid + k * NTHREADS;
      const int b = idx >> 7, u = idx & 127;
      const float zi = zbuf[b * ZPITCH + u];
      const float zf = zbuf[b * ZPITCH + UNITS + u];
      const float zo = zbuf[b * ZPITCH + 2 * UNITS + u];
      const float zu = zbuf[b * ZPITCH + 3 * UNITS + u];
      const float ig = fast_sigmoid(zi);
      const float fg = fast_sigmoid(zf);
      const float og = fast_sigmoid(zo);
      const float ug = fast_tanh(zu);
      const float c  = fg * c_reg[k] + ig * ug;
      const float h  = og * fast_tanh(c);
      c_reg[k] = c;
      h_reg[k] = h;
      hbuf16[b * (2 * APITCH) + u] = f2bf(h);
      out[((size_t)t * BATCH + b0 + b) * UNITS + u] = h;
    }
#if USE_TDM
    if (wave == 0) wait_tensorcnt_le1();   // tile for t+1 landed
#endif
    __syncthreads();   // publishes h, z reuse, and the t+1 x-tile
  }

  // ---- final stacked states [c; h] ---------------------------------------
  const size_t state_base = (size_t)T_STEPS * BATCH * UNITS;
#pragma unroll
  for (int k = 0; k < 4; ++k) {
    const int idx = tid + k * NTHREADS;
    const int b = idx >> 7, u = idx & 127;
    out[state_base + (b0 + b) * UNITS + u]                 = c_reg[k];
    out[state_base + BATCH * UNITS + (b0 + b) * UNITS + u] = h_reg[k];
  }
}

extern "C" void kernel_launch(void* const* d_in, const int* in_sizes, int n_in,
                              void* d_out, int out_size, void* d_ws, size_t ws_size,
                              hipStream_t stream) {
  (void)in_sizes; (void)n_in; (void)out_size; (void)d_ws; (void)ws_size;
  const float* inputs  = (const float*)d_in[0];
  const float* states  = (const float*)d_in[1];
  const float* dones   = (const float*)d_in[2];
  const float* kernelx = (const float*)d_in[3];
  const float* kernelh = (const float*)d_in[4];
  // d_in[5] = bias: created but unused by the reference
  float* out = (float*)d_out;

  dim3 grid(BATCH / BSLICE);   // 16 workgroups, one batch slice each
  dim3 block(NTHREADS);        // 16 wave32 waves
  lstm_scan_wmma<<<grid, block, LDS_BYTES, stream>>>(
      inputs, states, dones, kernelx, kernelh, out);
}
